// VectorQuantizer_88802743812542
// MI455X (gfx1250) — matrix operand order repaired and checked by reading, not compiled, not run on hardware
//
#include <hip/hip_runtime.h>

// CDNA5 / gfx1250 wave32 WMMA vector-quantizer.
// score = ||c||^2 - 2 x.c  (||x||^2 dropped: constant per row for argmin)
// GEMM done as split-bf16 (hi/lo) 3-product WMMA => ~f32 accuracy at bf16 rate.

typedef __bf16 v16bf __attribute__((ext_vector_type(16)));
typedef float  v8f   __attribute__((ext_vector_type(8)));

#define DIMS            256
#define CHUNKS          8        // DIMS / 32 (K per bf16 WMMA)
#define ROWS_PER_WAVE   16
#define WAVES_PER_BLOCK 8
#define ROWS_PER_BLOCK  (ROWS_PER_WAVE * WAVES_PER_BLOCK)

__device__ __forceinline__ void bf16_split(float f, unsigned short& hi, unsigned short& lo) {
  unsigned u = __builtin_bit_cast(unsigned, f);
  hi = (unsigned short)(u >> 16);                       // truncate -> hi
  float hf = __builtin_bit_cast(float, (unsigned)hi << 16);
  float lf = f - hf;                                    // residual, |lf| < 2^-7 |f|
  lo = (unsigned short)(__builtin_bit_cast(unsigned, lf) >> 16);
}

__device__ __forceinline__ __bf16 bits_to_bf16(unsigned short b) {
  return __builtin_bit_cast(__bf16, b);
}

// ---------------------------------------------------------------------------
// Prep: one wave per code. Compute ||c||^2 and split codebook into bf16 hi/lo.
// ---------------------------------------------------------------------------
__global__ __launch_bounds__(256)
void vq_prep(const float* __restrict__ cb, float* __restrict__ cnorm,
             unsigned short* __restrict__ cbh, unsigned short* __restrict__ cbl,
             int K) {
  const int lane = threadIdx.x & 31;
  const int code = (int)((blockIdx.x * blockDim.x + threadIdx.x) >> 5);
  if (code >= K) return;
  const float* cr = cb + (size_t)code * DIMS;
  float4 q0 = *(const float4*)(cr + lane * 8);
  float4 q1 = *(const float4*)(cr + lane * 8 + 4);
  float fv[8] = {q0.x, q0.y, q0.z, q0.w, q1.x, q1.y, q1.z, q1.w};
  unsigned hp[4], lp[4];
  float s = 0.f;
#pragma unroll
  for (int e = 0; e < 4; ++e) {
    unsigned short h0, l0, h1, l1;
    bf16_split(fv[2 * e],     h0, l0);
    bf16_split(fv[2 * e + 1], h1, l1);
    hp[e] = (unsigned)h0 | ((unsigned)h1 << 16);
    lp[e] = (unsigned)l0 | ((unsigned)l1 << 16);
    s += fv[2 * e] * fv[2 * e] + fv[2 * e + 1] * fv[2 * e + 1];
  }
  *(uint4*)(cbh + (size_t)code * DIMS + lane * 8) = make_uint4(hp[0], hp[1], hp[2], hp[3]);
  *(uint4*)(cbl + (size_t)code * DIMS + lane * 8) = make_uint4(lp[0], lp[1], lp[2], lp[3]);
#pragma unroll
  for (int off = 16; off > 0; off >>= 1) s += __shfl_xor(s, off, 32);
  if (lane == 0) cnorm[code] = s;
}

// ---------------------------------------------------------------------------
// Main: 8 waves/block, 16 rows/wave. A tile (16x256) resident as bf16 hi/lo
// fragments; stream codebook in 16-code tiles; 3 WMMAs per 32-wide K chunk.
// C layout (16x16 f32): VGPR r -> rows r (lanes 0-15) / r+8 (lanes 16-31),
// column = lane&15  =>  per-lane running argmin, 16-lane butterfly at the end.
// ---------------------------------------------------------------------------
template <bool PRE>
__global__ __launch_bounds__(256)
void vq_argmin(const float* __restrict__ x,
               const float* __restrict__ cb,
               const unsigned short* __restrict__ cbh,
               const unsigned short* __restrict__ cbl,
               const float* __restrict__ cnorm,
               float* __restrict__ out,
               int ktiles) {
  const int lane = threadIdx.x & 31;
  const int wave = threadIdx.x >> 5;
  const int h    = lane >> 4;   // lane half: selects K sub-range per ISA layout
  const int nl   = lane & 15;
  const size_t rowbase = (size_t)blockIdx.x * ROWS_PER_BLOCK + (size_t)wave * ROWS_PER_WAVE;

  // ---- Load this wave's A tile once; split each element into bf16 hi/lo.
  // A k order chosen here (B uses the same): lane -> row M = lane&15; elems 0..15 -> K = h*16+0..15
  // (per chunk of 32).
  v16bf a_hi[CHUNKS], a_lo[CHUNKS];
  {
    const float* xr = x + (rowbase + (size_t)nl) * DIMS;
#pragma unroll
    for (int c = 0; c < CHUNKS; ++c) {
#pragma unroll
      for (int g = 0; g < 2; ++g) {
        const float* p = xr + c * 32 + h * 16 + g * 8;
        float4 q0 = *(const float4*)(p);
        float4 q1 = *(const float4*)(p + 4);
        float fv[8] = {q0.x, q0.y, q0.z, q0.w, q1.x, q1.y, q1.z, q1.w};
#pragma unroll
        for (int e = 0; e < 8; ++e) {
          unsigned short hb, lb;
          bf16_split(fv[e], hb, lb);
          a_hi[c][g * 8 + e] = bits_to_bf16(hb);
          a_lo[c][g * 8 + e] = bits_to_bf16(lb);
        }
      }
    }
  }

  float bestd[8];
  int   besti[8];
#pragma unroll
  for (int r = 0; r < 8; ++r) { bestd[r] = __builtin_inff(); besti[r] = 0; }

  for (int t = 0; t < ktiles; ++t) {
    const int code = t * 16 + nl;
    v8f acc = {0.f, 0.f, 0.f, 0.f, 0.f, 0.f, 0.f, 0.f};
    float cn;

    if constexpr (PRE) {
      if (t + 1 < ktiles)  // uniform branch; speculative prefetch of next tile
        __builtin_prefetch(cbh + ((size_t)code + 16) * DIMS, 0, 0);
      cn = cnorm[code];
#pragma unroll
      for (int c = 0; c < CHUNKS; ++c) {
        // B k order chosen here (same as A): lane -> col N = lane&15; elems -> K = h*16+0..15
        const size_t boff = (size_t)code * DIMS + c * 32 + h * 16;
        v16bf bh = *(const v16bf*)(cbh + boff);
        v16bf bl = *(const v16bf*)(cbl + boff);
        acc = __builtin_amdgcn_wmma_f32_16x16x32_bf16(false, a_hi[c], false, bh, (short)0, acc, false, false);
        acc = __builtin_amdgcn_wmma_f32_16x16x32_bf16(false, a_hi[c], false, bl, (short)0, acc, false, false);
        acc = __builtin_amdgcn_wmma_f32_16x16x32_bf16(false, a_lo[c], false, bh, (short)0, acc, false, false);
      }
    } else {
      // Fallback: convert f32 codebook inline; ||c||^2 built from the same
      // loads (each lane holds half the dims; partner lane l^16 has the rest).
      float sq = 0.f;
#pragma unroll
      for (int c = 0; c < CHUNKS; ++c) {
        const float* p = cb + (size_t)code * DIMS + c * 32 + h * 16;
        float4 q0 = *(const float4*)(p);
        float4 q1 = *(const float4*)(p + 4);
        float4 q2 = *(const float4*)(p + 8);
        float4 q3 = *(const float4*)(p + 12);
        float fv[16] = {q0.x, q0.y, q0.z, q0.w, q1.x, q1.y, q1.z, q1.w,
                        q2.x, q2.y, q2.z, q2.w, q3.x, q3.y, q3.z, q3.w};
        v16bf bh, bl;
#pragma unroll
        for (int e = 0; e < 16; ++e) {
          unsigned short hb, lb;
          bf16_split(fv[e], hb, lb);
          bh[e] = bits_to_bf16(hb);
          bl[e] = bits_to_bf16(lb);
          sq += fv[e] * fv[e];
        }
        acc = __builtin_amdgcn_wmma_f32_16x16x32_bf16(false, a_hi[c], false, bh, (short)0, acc, false, false);
        acc = __builtin_amdgcn_wmma_f32_16x16x32_bf16(false, a_hi[c], false, bl, (short)0, acc, false, false);
        acc = __builtin_amdgcn_wmma_f32_16x16x32_bf16(false, a_lo[c], false, bh, (short)0, acc, false, false);
      }
      cn = sq + __shfl_xor(sq, 16, 32);  // combine the two K halves
    }

#pragma unroll
    for (int r = 0; r < 8; ++r) {
      float score = cn - 2.f * acc[r];
      if (score < bestd[r]) { bestd[r] = score; besti[r] = code; }  // '<' keeps lowest idx
    }
  }

  // Cross-lane argmin within each 16-lane half (jnp tie-break: lowest index).
#pragma unroll
  for (int r = 0; r < 8; ++r) {
    float d = bestd[r];
    int   i = besti[r];
#pragma unroll
    for (int off = 8; off > 0; off >>= 1) {
      float od = __shfl_xor(d, off, 16);
      int   oi = __shfl_xor(i, off, 16);
      if (od < d || (od == d && oi < i)) { d = od; i = oi; }
    }
    if (nl == 0) out[rowbase + (size_t)(h * 8 + r)] = (float)i;
  }
}

// ---------------------------------------------------------------------------
extern "C" void kernel_launch(void* const* d_in, const int* in_sizes, int n_in,
                              void* d_out, int out_size, void* d_ws, size_t ws_size,
                              hipStream_t stream) {
  const float* latents  = (const float*)d_in[0];  // [8,8,512,256] f32
  const float* codebook = (const float*)d_in[1];  // [8192,256]    f32
  float* out = (float*)d_out;                     // [32768] indices as f32

  const int N = in_sizes[0] / DIMS;  // 32768
  const int K = in_sizes[1] / DIMS;  // 8192
  const int ktiles  = K / 16;
  const int nblocks = N / ROWS_PER_BLOCK;

  const size_t need = (size_t)K * sizeof(float)                    // cnorm
                    + 2 * (size_t)K * DIMS * sizeof(unsigned short); // hi + lo
  if (d_ws && ws_size >= need) {
    float*          cnorm = (float*)d_ws;
    unsigned short* cbh   = (unsigned short*)((char*)d_ws + (size_t)K * sizeof(float));
    unsigned short* cbl   = cbh + (size_t)K * DIMS;
    vq_prep<<<(K + 7) / 8, 256, 0, stream>>>(codebook, cnorm, cbh, cbl, K);
    vq_argmin<true><<<nblocks, 256, 0, stream>>>(latents, codebook, cbh, cbl, cnorm, out, ktiles);
  } else {
    vq_argmin<false><<<nblocks, 256, 0, stream>>>(latents, codebook, nullptr, nullptr, nullptr, out, ktiles);
  }
}
